// MultiLayerRNN_30159260352547
// MI455X (gfx1250) — matrix operand order repaired and checked by reading, not compiled, not run on hardware
//
#include <hip/hip_runtime.h>
#include <hip/hip_bf16.h>

// ---------------------------------------------------------------------------
// 2-layer LSTM LM for MI455X (gfx1250, wave32, WMMA bf16 16x16x32)
//   B=8 T=512 V=32000 E=256 H=1024, gates 4H=4096, order f,i,o,c_hat (all sigmoid)
// ---------------------------------------------------------------------------

#define B_ 8
#define T_ 512
#define V_ 32000
#define E_ 256
#define H_ 1024
#define G_ 4096   // 4*H

typedef __attribute__((ext_vector_type(16))) __bf16        v16bf;
typedef __attribute__((ext_vector_type(8)))  float         v8f;
typedef __attribute__((ext_vector_type(4)))  unsigned int  v4u;
typedef unsigned short u16;

union ABFrag { v16bf v; v4u q[2]; };

// CDNA5 async global->LDS path (ASYNCcnt), guarded so the file compiles on
// toolchains that do not declare the builtins (falls back to sync staging).
#if __has_builtin(__builtin_amdgcn_global_load_async_to_lds_b128) && \
    __has_builtin(__builtin_amdgcn_s_wait_asynccnt)
#define USE_ASYNC_LDS 1
#else
#define USE_ASYNC_LDS 0
#endif

#define AS1 __attribute__((address_space(1)))
#define AS3 __attribute__((address_space(3)))

// Builtin parameter type per hipcc diagnostic:
//   'int __attribute__((vector_size(16))) __device__ *'  (AS1, non-const)
typedef int v4i_vs __attribute__((vector_size(16)));

// Global generic pointer is numerically the as(1) address; generic LDS
// pointer's low 32 bits are the LDS byte offset (flat aperture rule), so
// integer casts yield valid AS-qualified pointers.
__device__ __forceinline__ AS1 v4i_vs* gl_cast_v(const void* p) {
  return (AS1 v4i_vs*)(unsigned long long)p;
}
__device__ __forceinline__ AS3 v4i_vs* lds_cast_v(void* p) {
  return (AS3 v4i_vs*)(unsigned int)(unsigned long long)p;
}

__device__ __forceinline__ u16 f32_to_bf16(float f) {
  unsigned int u = __float_as_uint(f);
  u += 0x7fffu + ((u >> 16) & 1u);      // round-to-nearest-even
  return (u16)(u >> 16);
}

__device__ __forceinline__ v8f wmma_bf16(const ABFrag& a, const ABFrag& b, v8f c) {
  // D = A(16x32 bf16) * B(32x16 bf16) + C(16x16 f32)
  return __builtin_amdgcn_wmma_f32_16x16x32_bf16(false, a.v, false, b.v,
                                                 (short)0, c, false, false);
}

// ---------------------------------------------------------------------------
// Weight prep: f32 [K,N] -> bf16 [N,K]  (coalesced both ways via LDS tile)
// ---------------------------------------------------------------------------
__global__ __launch_bounds__(256) void transpose_f32_bf16(
    const float* __restrict__ W, u16* __restrict__ WT, int K, int N)
{
  __shared__ float tile[32][33];
  const int kb = blockIdx.y * 32, nb = blockIdx.x * 32;
  const int tx = threadIdx.x, ty = threadIdx.y;      // block (32,8)
  #pragma unroll
  for (int i = 0; i < 32; i += 8)
    tile[ty + i][tx] = W[(size_t)(kb + ty + i) * N + (nb + tx)];
  __syncthreads();
  #pragma unroll
  for (int i = 0; i < 32; i += 8)
    WT[(size_t)(nb + ty + i) * K + (kb + tx)] = f32_to_bf16(tile[tx][ty + i]);
}

// ---------------------------------------------------------------------------
// Embedding gather -> bf16 A matrix [B*T, E]
// ---------------------------------------------------------------------------
__global__ __launch_bounds__(256) void embed_bf16(
    const int* __restrict__ x, const float* __restrict__ emb,
    u16* __restrict__ xe)
{
  const int total = B_ * T_ * E_;
  for (int i = blockIdx.x * blockDim.x + threadIdx.x; i < total;
       i += gridDim.x * blockDim.x) {
    const int r = i >> 8;              // row b*T+t
    const int e = i & (E_ - 1);
    xe[i] = f32_to_bf16(emb[(size_t)x[r] * E_ + e]);
  }
}

// ---------------------------------------------------------------------------
// GEMM: C[M,N] f32 = A_bf16[M,K] * BT_bf16[N,K] + bias[N]
// block tile 128x128, 8 waves (4 along M x 2 along N), each wave 2x4 16x16
// tiles. Double-buffered LDS; next K-tile staged with async global->LDS
// (ASYNCcnt) overlapping the WMMAs on the current tile.
// M%128==0, N%128==0, K%32==0 (holds for all uses)
// ---------------------------------------------------------------------------
__global__ __launch_bounds__(256) void gemm_bf16_tn(
    const u16* __restrict__ A, const u16* __restrict__ BT,
    const float* __restrict__ bias, float* __restrict__ C,
    int M, int N, int K)
{
  __shared__ __align__(16) u16 As[2][128 * 32];
  __shared__ __align__(16) u16 Bs[2][128 * 32];

  const int tid  = threadIdx.x;
  const int lane = tid & 31;
  const int wave = tid >> 5;
  const int m0 = blockIdx.y * 128, n0 = blockIdx.x * 128;
  const int wmOff = (wave >> 1) * 32;   // 4 waves along M, 32 rows each
  const int wnOff = (wave & 1) * 64;    // 2 waves along N, 64 cols each

  const int l15   = lane & 15;
  const int halfm = lane >> 4;          // 0: lanes 0-15, 1: lanes 16-31
  const int kbA   = halfm * 16;         // A frag: K 0-15 vs 16-31
  const int kbB   = halfm * 16;         // B frag: K 0-15 vs 16-31

  v8f acc[2][4];
  v8f vzero = {};
  #pragma unroll
  for (int i = 0; i < 2; ++i)
    #pragma unroll
    for (int j = 0; j < 4; ++j) acc[i][j] = vzero;

  const int lr = tid >> 1;              // tile row 0..127 (2 threads per row)
  const int lc = (tid & 1) * 16;        // element-col 0 or 16
  const size_t aRow = (size_t)(m0 + lr) * K;
  const size_t bRow = (size_t)(n0 + lr) * K;
  const int nk = K >> 5;                // number of 32-wide K tiles

  // stage K-tile i of A and BT into LDS buffer p (4 x 16B per thread)
  auto issue_tile = [&](int i, int p) {
    const u16* ga = &A [aRow + (size_t)i * 32 + lc];
    const u16* gb = &BT[bRow + (size_t)i * 32 + lc];
    u16* la = &As[p][lr * 32 + lc];
    u16* lb = &Bs[p][lr * 32 + lc];
#if USE_ASYNC_LDS
    __builtin_amdgcn_global_load_async_to_lds_b128(gl_cast_v(ga),     lds_cast_v(la),     0, 0);
    __builtin_amdgcn_global_load_async_to_lds_b128(gl_cast_v(ga + 8), lds_cast_v(la + 8), 0, 0);
    __builtin_amdgcn_global_load_async_to_lds_b128(gl_cast_v(gb),     lds_cast_v(lb),     0, 0);
    __builtin_amdgcn_global_load_async_to_lds_b128(gl_cast_v(gb + 8), lds_cast_v(lb + 8), 0, 0);
#else
    *(v4u*)la       = *(const v4u*)ga;
    *(v4u*)(la + 8) = *(const v4u*)(ga + 8);
    *(v4u*)lb       = *(const v4u*)gb;
    *(v4u*)(lb + 8) = *(const v4u*)(gb + 8);
#endif
  };

  issue_tile(0, 0);

  for (int i = 0; i < nk; ++i) {
    const int p = i & 1;
    if (i + 1 < nk) {
      issue_tile(i + 1, p ^ 1);         // overlap with compute on tile i
#if USE_ASYNC_LDS
      __builtin_amdgcn_s_wait_asynccnt(4);   // tile i's 4 asyncs complete
#else
      __builtin_prefetch(&A [aRow + (size_t)(i + 1) * 32 + lc], 0, 1);
      __builtin_prefetch(&BT[bRow + (size_t)(i + 1) * 32 + lc], 0, 1);
#endif
    } else {
#if USE_ASYNC_LDS
      __builtin_amdgcn_s_wait_asynccnt(0);   // final tile complete
#endif
    }
    __syncthreads();                    // all waves' tile-i data visible

    ABFrag af[2];
    #pragma unroll
    for (int mt = 0; mt < 2; ++mt) {
      const u16* s = &As[p][(wmOff + mt * 16 + l15) * 32 + kbA];
      af[mt].q[0] = *(const v4u*)(s);
      af[mt].q[1] = *(const v4u*)(s + 8);
    }
    ABFrag bfr[4];
    #pragma unroll
    for (int nt = 0; nt < 4; ++nt) {
      const u16* s = &Bs[p][(wnOff + nt * 16 + l15) * 32 + kbB];
      bfr[nt].q[0] = *(const v4u*)(s);
      bfr[nt].q[1] = *(const v4u*)(s + 8);
    }
    #pragma unroll
    for (int mt = 0; mt < 2; ++mt)
      #pragma unroll
      for (int nt = 0; nt < 4; ++nt)
        acc[mt][nt] = wmma_bf16(af[mt], bfr[nt], acc[mt][nt]);

    __syncthreads();                    // done reading buf p before reuse
  }

  // store: D layout — lane l, vgpr r -> row r + 8*(l>>4), col l&15
  #pragma unroll
  for (int mt = 0; mt < 2; ++mt) {
    #pragma unroll
    for (int nt = 0; nt < 4; ++nt) {
      const int col = n0 + wnOff + nt * 16 + l15;
      const float bv = bias ? bias[col] : 0.0f;
      #pragma unroll
      for (int r = 0; r < 8; ++r) {
        const int row = m0 + wmOff + mt * 16 + r + 8 * halfm;
        C[(size_t)row * N + col] = acc[mt][nt][r] + bv;
      }
    }
  }
}

// ---------------------------------------------------------------------------
// State init: c=[16,H] f32, h ping/pong bf16 [16,H] (rows 8..15 zero pad)
// ---------------------------------------------------------------------------
__global__ __launch_bounds__(256) void init_state(
    const float* __restrict__ c0, const float* __restrict__ h0,
    float* __restrict__ c, u16* __restrict__ h_a, u16* __restrict__ h_b)
{
  const int j = blockIdx.x * blockDim.x + threadIdx.x;
  if (j >= 16 * H_) return;
  const int m = j >> 10, col = j & (H_ - 1);
  const float cv = (m < B_) ? c0[col] : 0.0f;
  const float hv = (m < B_) ? h0[col] : 0.0f;
  c[j] = cv;
  const u16 hb = f32_to_bf16(hv);
  h_a[j] = hb;
  h_b[j] = hb;
}

// ---------------------------------------------------------------------------
// One LSTM time step. Grid = H/64 blocks; block owns h columns [j0, j0+64)
// and the 4 matching gate column groups (16 N-tiles of 16). 8 waves x 2 tiles.
//   gates = sigmoid(xproj[:,t] + h_in @ whT^T)   (bias folded into xproj)
//   c = f*c + i*chat ; h_out = o*tanh(c) ; yseq = o
// ---------------------------------------------------------------------------
__global__ __launch_bounds__(256) void lstm_step(
    const u16* __restrict__ h_in, u16* __restrict__ h_out,
    float* __restrict__ cst, const u16* __restrict__ whT,
    const float* __restrict__ xproj, u16* __restrict__ yseq, int t)
{
  __shared__ float glds[4][8][64];      // [gate][batch][local h col]
  const int tid   = threadIdx.x;
  const int lane  = tid & 31;
  const int wave  = tid >> 5;
  const int l15   = lane & 15;
  const int halfm = lane >> 4;
  const int j0    = blockIdx.x * 64;

  v8f acc[2];
  v8f vzero = {};
  acc[0] = vzero; acc[1] = vzero;

  int ncol[2], qg[2], sg[2];
  #pragma unroll
  for (int i = 0; i < 2; ++i) {
    const int tIdx = wave * 2 + i;      // 0..15 within block
    qg[i] = tIdx >> 2;                  // gate id: 0=f 1=i 2=o 3=chat
    sg[i] = tIdx & 3;                   // 16-col subtile within the 64 cols
    ncol[i] = qg[i] * H_ + j0 + sg[i] * 16;
  }

  for (int k0 = 0; k0 < H_; k0 += 32) {
    ABFrag af;                          // A = h_in bf16 [16, H], row = l15
    {
      const u16* p = &h_in[(size_t)l15 * H_ + k0 + halfm * 16];
      af.q[0] = *(const v4u*)(p);
      af.q[1] = *(const v4u*)(p + 8);
    }
    #pragma unroll
    for (int i = 0; i < 2; ++i) {
      ABFrag bfr;                       // B = whT bf16 [4H, H], row = gate col
      const u16* p = &whT[(size_t)(ncol[i] + l15) * H_ + k0 + halfm * 16];
      bfr.q[0] = *(const v4u*)(p);
      bfr.q[1] = *(const v4u*)(p + 8);
      acc[i] = wmma_bf16(af, bfr, acc[i]);
    }
  }

  // sigmoid(acc + xproj) for real batch rows (m<8 live in lanes 0..15)
  #pragma unroll
  for (int i = 0; i < 2; ++i) {
    if (halfm == 0) {
      #pragma unroll
      for (int r = 0; r < 8; ++r) {
        const int m = r;                              // batch row
        const int n = ncol[i] + l15;                  // gate column
        float g = acc[i][r] + xproj[((size_t)m * T_ + t) * G_ + n];
        g = 1.0f / (1.0f + __expf(-g));
        glds[qg[i]][m][sg[i] * 16 + l15] = g;
      }
    }
  }
  __syncthreads();

  // elementwise cell/hidden update on this block's 64 h columns
  for (int idx = tid; idx < 8 * 64; idx += 256) {
    const int m  = idx >> 6;
    const int jl = idx & 63;
    const int j  = j0 + jl;
    const float f  = glds[0][m][jl];
    const float ig = glds[1][m][jl];
    const float o  = glds[2][m][jl];
    const float ch = glds[3][m][jl];
    float c = f * cst[(size_t)m * H_ + j] + ig * ch;
    cst[(size_t)m * H_ + j] = c;
    h_out[(size_t)m * H_ + j] = f32_to_bf16(o * tanhf(c));
    yseq[((size_t)m * T_ + t) * H_ + j] = f32_to_bf16(o);   // layer output = o
  }
}

// ---------------------------------------------------------------------------
// Second tuple output: logits[:, -1]
// ---------------------------------------------------------------------------
__global__ __launch_bounds__(256) void last_logits(
    const float* __restrict__ logits, float* __restrict__ out2)
{
  const int i = blockIdx.x * blockDim.x + threadIdx.x;
  if (i >= B_ * V_) return;
  const int b = i / V_, v = i % V_;
  out2[i] = logits[((size_t)b * T_ + (T_ - 1)) * V_ + v];
}

// ---------------------------------------------------------------------------
// Host-side orchestration
// ---------------------------------------------------------------------------
extern "C" void kernel_launch(void* const* d_in, const int* in_sizes, int n_in,
                              void* d_out, int out_size, void* d_ws, size_t ws_size,
                              hipStream_t stream)
{
  (void)in_sizes; (void)n_in; (void)out_size; (void)ws_size;
  const int*   x     = (const int*)  d_in[0];
  const float* emb   = (const float*)d_in[1];
  const float* wx0   = (const float*)d_in[2];
  const float* wh0   = (const float*)d_in[3];
  const float* bh0   = (const float*)d_in[4];
  const float* c00   = (const float*)d_in[5];
  const float* h00   = (const float*)d_in[6];
  const float* wx1   = (const float*)d_in[7];
  const float* wh1   = (const float*)d_in[8];
  const float* bh1   = (const float*)d_in[9];
  const float* c01   = (const float*)d_in[10];
  const float* h01   = (const float*)d_in[11];
  const float* headw = (const float*)d_in[12];
  const float* headb = (const float*)d_in[13];
  float* out = (float*)d_out;

  // workspace carve-out (256B aligned)
  char* p = (char*)d_ws;
  auto take = [&p](size_t bytes) -> char* {
    char* r = p; p += (bytes + 255) & ~(size_t)255; return r;
  };
  u16*   wx0T  = (u16*)  take((size_t)G_ * E_ * 2);      //  2 MB
  u16*   wh0T  = (u16*)  take((size_t)G_ * H_ * 2);      //  8 MB
  u16*   wx1T  = (u16*)  take((size_t)G_ * H_ * 2);      //  8 MB
  u16*   wh1T  = (u16*)  take((size_t)G_ * H_ * 2);      //  8 MB
  u16*   headT = (u16*)  take((size_t)V_ * H_ * 2);      // 64 MB
  u16*   xe    = (u16*)  take((size_t)B_ * T_ * E_ * 2); //  2 MB
  float* xproj = (float*)take((size_t)B_ * T_ * G_ * 4); // 64 MB (reused L0/L1)
  u16*   y0    = (u16*)  take((size_t)B_ * T_ * H_ * 2); //  8 MB
  u16*   y1    = (u16*)  take((size_t)B_ * T_ * H_ * 2); //  8 MB
  u16*   h0a   = (u16*)  take((size_t)16 * H_ * 2);
  u16*   h0b   = (u16*)  take((size_t)16 * H_ * 2);
  u16*   h1a   = (u16*)  take((size_t)16 * H_ * 2);
  u16*   h1b   = (u16*)  take((size_t)16 * H_ * 2);
  float* c0buf = (float*)take((size_t)16 * H_ * 4);
  float* c1buf = (float*)take((size_t)16 * H_ * 4);

  const dim3 blk256(256), blkT(32, 8);

  // 1) weight prep: bf16 transposed [N,K]
  transpose_f32_bf16<<<dim3(G_ / 32, E_ / 32), blkT, 0, stream>>>(wx0, wx0T, E_, G_);
  transpose_f32_bf16<<<dim3(G_ / 32, H_ / 32), blkT, 0, stream>>>(wh0, wh0T, H_, G_);
  transpose_f32_bf16<<<dim3(G_ / 32, H_ / 32), blkT, 0, stream>>>(wx1, wx1T, H_, G_);
  transpose_f32_bf16<<<dim3(G_ / 32, H_ / 32), blkT, 0, stream>>>(wh1, wh1T, H_, G_);
  transpose_f32_bf16<<<dim3(V_ / 32, H_ / 32), blkT, 0, stream>>>(headw, headT, H_, V_);

  // 2) embedding gather
  embed_bf16<<<dim3((B_ * T_ * E_) / 256), blk256, 0, stream>>>(x, emb, xe);

  // 3) layer 0: xproj0 = xe @ wx0 + bh0   [4096, 4096]
  gemm_bf16_tn<<<dim3(G_ / 128, (B_ * T_) / 128), blk256, 0, stream>>>(
      xe, wx0T, bh0, xproj, B_ * T_, G_, E_);

  init_state<<<dim3((16 * H_) / 256), blk256, 0, stream>>>(c00, h00, c0buf, h0a, h0b);
  {
    u16 *hin = h0a, *hout = h0b;
    for (int t = 0; t < T_; ++t) {
      lstm_step<<<dim3(H_ / 64), blk256, 0, stream>>>(hin, hout, c0buf, wh0T,
                                                      xproj, y0, t);
      u16* tmp = hin; hin = hout; hout = tmp;
    }
  }

  // 4) layer 1: xproj1 = y0 @ wx1 + bh1   [4096, 4096]
  gemm_bf16_tn<<<dim3(G_ / 128, (B_ * T_) / 128), blk256, 0, stream>>>(
      y0, wx1T, bh1, xproj, B_ * T_, G_, H_);

  init_state<<<dim3((16 * H_) / 256), blk256, 0, stream>>>(c01, h01, c1buf, h1a, h1b);
  {
    u16 *hin = h1a, *hout = h1b;
    for (int t = 0; t < T_; ++t) {
      lstm_step<<<dim3(H_ / 64), blk256, 0, stream>>>(hin, hout, c1buf, wh1T,
                                                      xproj, y1, t);
      u16* tmp = hin; hin = hout; hout = tmp;
    }
  }

  // 5) head: logits = y1 @ head_w + head_b   [4096, 32000] -> d_out
  gemm_bf16_tn<<<dim3(V_ / 128, (B_ * T_) / 128), blk256, 0, stream>>>(
      y1, headT, headb, out, B_ * T_, V_, H_);

  // 6) logits[:, -1] -> second tuple output
  last_logits<<<dim3((B_ * V_ + 255) / 256), blk256, 0, stream>>>(
      out, out + (size_t)B_ * T_ * V_);
}
